// KVCacheAttention_20864951124059
// MI455X (gfx1250) — matrix operand order repaired and checked by reading, not compiled, not run on hardware
//
#include <hip/hip_runtime.h>

#define D_MODEL 2048
#define NH 16
#define DH 128
#define SB 2          // batch
#define SS 2048       // new seq len
#define PP 2048       // past len
#define TT 4096       // total keys
#define SCALE 0.08838834764831845f

typedef __attribute__((ext_vector_type(16))) __bf16 v16bf;
typedef __attribute__((ext_vector_type(8)))  __bf16 v8bf;
typedef __attribute__((ext_vector_type(4)))  __bf16 v4bf;
typedef __attribute__((ext_vector_type(8)))  float  v8f;
typedef __attribute__((ext_vector_type(4)))  float  v4f;

// native hardware convert (v_cvt_*bf16*), RNE
__device__ __forceinline__ __bf16 f2bf(float f) { return (__bf16)f; }

__device__ __forceinline__ v8f wmma_bf16(v16bf a, v16bf b, v8f c) {
  return __builtin_amdgcn_wmma_f32_16x16x32_bf16(
      /*neg_a=*/false, a, /*neg_b=*/false, b,
      /*c_mod=*/(short)0, c, /*reuse_a=*/false, /*reuse_b=*/false);
}

// A-fragment (16x32 bf16, M x K): lane holds row M=lane&15; K=(lane>>4)*16+e
// contiguous (khalf=(lane>>4)*16).
__device__ __forceinline__ v16bf fragA(const __bf16* rowp, int khalf) {
  v8bf lo = *(const v8bf*)(rowp + khalf);
  v8bf hi = *(const v8bf*)(rowp + khalf + 8);
  v16bf f;
#pragma unroll
  for (int i = 0; i < 8; ++i) { f[i] = lo[i]; f[i + 8] = hi[i]; }
  return f;
}

// B-fragment (32x16 bf16, K x N): lane col N=lane&15; K=(lane>>4)*16+e contiguous.
__device__ __forceinline__ v16bf fragB(const __bf16* rowp, int kbase) {
  v8bf lo = *(const v8bf*)(rowp + kbase);
  v8bf hi = *(const v8bf*)(rowp + kbase + 8);
  v16bf f;
#pragma unroll
  for (int i = 0; i < 8; ++i) { f[i] = lo[i]; f[i + 8] = hi[i]; }
  return f;
}

// async 16B global -> LDS copy (GVS mode: saddr base + per-lane u32 offset)
__device__ __forceinline__ void async_b128(unsigned lds_addr, unsigned goff,
                                           const void* base) {
  asm volatile("global_load_async_to_lds_b128 %0, %1, %2"
               :: "v"(lds_addr), "v"(goff),
                  "s"((unsigned long long)base) : "memory");
}
__device__ __forceinline__ void wait_async0() {
  asm volatile("s_wait_asynccnt 0x0" ::: "memory");
}

// ---------------------------------------------------------------------------
// past K/V -> front of fp32 cache in d_out, plus bf16 K and transposed bf16 V
// staging buffers for the attention kernel.
__global__ __launch_bounds__(256)
void copy_past(const v4f* __restrict__ pk, const v4f* __restrict__ pv,
               v4f* __restrict__ ko, v4f* __restrict__ vo,
               __bf16* __restrict__ kbf, __bf16* __restrict__ vbfT) {
  size_t i = (size_t)blockIdx.x * 256 + threadIdx.x;   // 2,097,152 v4f total
  size_t bh  = i >> 16;          // 65536 v4f per (b,h) source block
  size_t rem = i & 65535;
  size_t d = bh * 131072 + rem;  // dst block stride = 4096*128/4
  v4f kk = pk[i];
  v4f vv = pv[i];
  ko[d] = kk;
  vo[d] = vv;
  size_t s   = rem >> 5;         // key row 0..2047
  size_t dh4 = (rem & 31) * 4;   // dh group
  v4bf kb;
#pragma unroll
  for (int q = 0; q < 4; ++q) kb[q] = f2bf(kk[q]);
  *(v4bf*)(kbf + (bh * TT + s) * DH + dh4) = kb;
#pragma unroll
  for (int q = 0; q < 4; ++q)
    vbfT[(bh * DH + dh4 + q) * TT + s] = f2bf(vv[q]);
}

// ---------------------------------------------------------------------------
// C[m,n] = sum_k A[m,k] * W[n,k] + bias[n]   (y = x @ W^T + b)
// AKIND 0: A fp32 (convert on LDS fill); AKIND 1: A bf16 (direct fill).
// mode 0: fp32 row-major      -> dstF            (output projection)
// mode 1: bf16 row-major      -> dstB            (Q staging)
// mode 2: fp32 K-cache tail   -> dstF, bf16 Kbf  -> dstB (same layout)
// mode 3: fp32 V-cache tail   -> dstF, bf16 V^T  -> dstB ([bh][dh][key])
#define GSTR 40   // LDS row stride (halves): 16B-aligned, bank-conflict-free
template<int AKIND>
__global__ __launch_bounds__(256)
void gemm_wmma(const void* __restrict__ A, const float* __restrict__ W,
               const float* __restrict__ bias, float* __restrict__ dstF,
               __bf16* __restrict__ dstB, int mode) {
  __shared__ __align__(16) __bf16 At[128 * GSTR];
  __shared__ __align__(16) __bf16 Bt[128 * GSTR];
  const int tid  = threadIdx.x;
  const int lane = tid & 31, wave = tid >> 5;
  const int wm = wave & 1, wn = wave >> 1;          // 2 x 4 wave grid
  const int n0 = blockIdx.x * 128, m0 = blockIdx.y * 128;
  const int l15 = lane & 15, lhi = lane >> 4;

  v8f c[4][2];
#pragma unroll
  for (int i = 0; i < 4; ++i)
#pragma unroll
    for (int j = 0; j < 2; ++j)
#pragma unroll
      for (int r = 0; r < 8; ++r) c[i][j][r] = 0.0f;

  for (int k0 = 0; k0 < D_MODEL; k0 += 32) {
    // W tile: 128x32 fp32 -> bf16
#pragma unroll
    for (int it = 0; it < 4; ++it) {
      int idx = tid + it * 256;        // 8 v4f per 32-wide row
      int row = idx >> 3;
      int c4  = (idx & 7) * 4;
      v4f wv = *(const v4f*)(W + (size_t)(n0 + row) * D_MODEL + k0 + c4);
      v4bf wb;
#pragma unroll
      for (int q = 0; q < 4; ++q) wb[q] = f2bf(wv[q]);
      *(v4bf*)(Bt + row * GSTR + c4) = wb;
    }
    // A tile
    if (AKIND == 0) {
      const float* Af = (const float*)A;
#pragma unroll
      for (int it = 0; it < 4; ++it) {
        int idx = tid + it * 256;
        int row = idx >> 3;
        int c4  = (idx & 7) * 4;
        v4f av = *(const v4f*)(Af + (size_t)(m0 + row) * D_MODEL + k0 + c4);
        v4bf ab;
#pragma unroll
        for (int q = 0; q < 4; ++q) ab[q] = f2bf(av[q]);
        *(v4bf*)(At + row * GSTR + c4) = ab;
      }
    } else {
      const __bf16* Ab = (const __bf16*)A;
#pragma unroll
      for (int it = 0; it < 2; ++it) {
        int idx = tid + it * 256;      // 4 v8bf per 32-wide row
        int row = idx >> 2;
        int g8  = (idx & 3) * 8;
        *(v8bf*)(At + row * GSTR + g8) =
            *(const v8bf*)(Ab + (size_t)(m0 + row) * D_MODEL + k0 + g8);
      }
    }
    __syncthreads();

    v16bf bfr[2];
#pragma unroll
    for (int j = 0; j < 2; ++j)
      bfr[j] = fragB(Bt + (wn * 32 + j * 16 + l15) * GSTR, lhi * 16);
#pragma unroll
    for (int i = 0; i < 4; ++i) {
      v16bf af = fragA(At + (wm * 64 + i * 16 + l15) * GSTR, lhi * 16);
#pragma unroll
      for (int j = 0; j < 2; ++j)
        c[i][j] = wmma_bf16(af, bfr[j], c[i][j]);
    }
    __syncthreads();
  }

  // epilogue (C/D layout: VGPR r -> row r + 8*(lane>=16), col lane&15)
#pragma unroll
  for (int i = 0; i < 4; ++i) {
#pragma unroll
    for (int j = 0; j < 2; ++j) {
      int col = n0 + wn * 32 + j * 16 + l15;
      float bv = bias[col];
#pragma unroll
      for (int r = 0; r < 8; ++r) {
        int row = m0 + wm * 64 + i * 16 + r + (lhi << 3);
        float v = c[i][j][r] + bv;
        if (mode == 0) {
          dstF[(size_t)row * D_MODEL + col] = v;
        } else if (mode == 1) {
          dstB[(size_t)row * D_MODEL + col] = f2bf(v);
        } else {
          int b = row >> 11, s = row & 2047;
          int h = col >> 7,  dh = col & 127;
          size_t ci = ((size_t)(b * NH + h) * TT + PP + s) * DH + dh;
          dstF[ci] = v;
          if (mode == 2) dstB[ci] = f2bf(v);
          else dstB[((size_t)(b * NH + h) * DH + dh) * TT + PP + s] = f2bf(v);
        }
      }
    }
  }
}

// ---------------------------------------------------------------------------
// Flash attention: block = 128 query rows of one (b,h); 8 waves x 16 rows.
// K/V chunks stream through LDS via global_load_async_to_lds_b128 (ASYNCcnt).
#define KCH 32
#define KSTR 136   // Kt[key][dh] row stride (halves); 272B, 16B-aligned
#define VSTR 40    // Vt[dh][key] row stride; 80B
#define PSTR 40
__global__ __launch_bounds__(256)
void attn_wmma(const __bf16* __restrict__ Qbf, const __bf16* __restrict__ Kbf,
               const __bf16* __restrict__ VbfT, __bf16* __restrict__ CtxBf) {
  __shared__ __align__(16) __bf16 Kt[KCH * KSTR];
  __shared__ __align__(16) __bf16 Vt[DH * VSTR];
  __shared__ __align__(16) __bf16 Pw[8 * 16 * PSTR];
  const int tid  = threadIdx.x;
  const int lane = tid & 31, wave = tid >> 5;
  const int l15 = lane & 15, lhi = lane >> 4;
  const int bh = blockIdx.y;
  const int b = bh >> 4, h = bh & 15;
  const int q0 = blockIdx.x * 128 + wave * 16;

  // Q fragments (16 rows x 128 dh), bf16 direct from staging buffer
  v16bf qf[4];
  const __bf16* qrow = Qbf + (size_t)(b * SS + q0 + l15) * D_MODEL + h * DH;
#pragma unroll
  for (int ch = 0; ch < 4; ++ch)
    qf[ch] = fragA(qrow + ch * 32, lhi * 16);

  v8f o[8];
  float m[8], l[8];
#pragma unroll
  for (int t = 0; t < 8; ++t)
#pragma unroll
    for (int r = 0; r < 8; ++r) o[t][r] = 0.0f;
#pragma unroll
  for (int r = 0; r < 8; ++r) { m[r] = -1e30f; l[r] = 0.0f; }

  const __bf16* kbase = Kbf + (size_t)bh * TT * DH;
  const __bf16* vbase = VbfT + (size_t)bh * DH * TT;

  for (int t0 = 0; t0 < TT; t0 += KCH) {
    // async fills: K chunk [32 key x 128 dh], V^T chunk [128 dh x 32 key]
    const __bf16* kchunk = kbase + (size_t)t0 * DH;   // 256B per key row
    const __bf16* vchunk = vbase + t0;                // 64B per dh row
#pragma unroll
    for (int it = 0; it < 2; ++it) {
      int idx = tid + it * 256;          // 0..511 16B transfers each
      int key = idx >> 4, o16 = idx & 15;
      async_b128((unsigned)(size_t)(void*)(Kt + key * KSTR) + o16 * 16,
                 (unsigned)(key * 256 + o16 * 16), kchunk);
      int dh = idx >> 2, o4 = idx & 3;
      async_b128((unsigned)(size_t)(void*)(Vt + dh * VSTR) + o4 * 16,
                 (unsigned)(dh * (TT * 2) + o4 * 16), vchunk);
    }
    wait_async0();
    __syncthreads();
    if (t0 + KCH < TT)                   // global_prefetch next K chunk
      __builtin_prefetch(kchunk + (size_t)KCH * DH + tid * 8, 0, 1);

    // scores: two 16-key subtiles, K-dim = dh (4 chunks of 32)
    v8f s[2];
#pragma unroll
    for (int st = 0; st < 2; ++st) {
#pragma unroll
      for (int r = 0; r < 8; ++r) s[st][r] = 0.0f;
#pragma unroll
      for (int ch = 0; ch < 4; ++ch) {
        v16bf kf = fragB(Kt + (st * 16 + l15) * KSTR + ch * 32, lhi * 16);
        s[st] = wmma_bf16(qf[ch], kf, s[st]);
      }
#pragma unroll
      for (int r = 0; r < 8; ++r) s[st][r] *= SCALE;
    }

    // online softmax; row stats reduced across 16 lanes of each half-wave
    float alpha[8];
#pragma unroll
    for (int r = 0; r < 8; ++r) {
      float mx = fmaxf(s[0][r], s[1][r]);
#pragma unroll
      for (int d = 8; d >= 1; d >>= 1) mx = fmaxf(mx, __shfl_xor(mx, d, 16));
      float mn = fmaxf(m[r], mx);
      alpha[r] = __expf(m[r] - mn);
      float p0 = __expf(s[0][r] - mn);
      float p1 = __expf(s[1][r] - mn);
      float rs = p0 + p1;
#pragma unroll
      for (int d = 8; d >= 1; d >>= 1) rs += __shfl_xor(rs, d, 16);
      l[r] = l[r] * alpha[r] + rs;
      m[r] = mn;
      // stage P (C-layout) into per-wave LDS scratch for A-layout reload
      __bf16* pr = Pw + (wave * 16 + r + (lhi << 3)) * PSTR;
      pr[l15]      = f2bf(p0);
      pr[16 + l15] = f2bf(p1);
    }
#pragma unroll
    for (int t = 0; t < 8; ++t)
#pragma unroll
      for (int r = 0; r < 8; ++r) o[t][r] *= alpha[r];

    // P @ V : P as A-frag (16x32), V chunk as B-frags over 8 dh tiles
    v16bf pf = fragA(Pw + (wave * 16 + l15) * PSTR, lhi * 16);
#pragma unroll
    for (int t = 0; t < 8; ++t) {
      v16bf vf = fragB(Vt + (t * 16 + l15) * VSTR, lhi * 16);
      o[t] = wmma_bf16(pf, vf, o[t]);
    }
    __syncthreads();
  }

  // normalize, write bf16 ctx in [B,S,H*Dh] layout for the output GEMM
#pragma unroll
  for (int r = 0; r < 8; ++r) {
    float inv = 1.0f / l[r];
    int srow = q0 + r + (lhi << 3);
    __bf16* crow = CtxBf + (size_t)(b * SS + srow) * D_MODEL + h * DH;
#pragma unroll
    for (int t = 0; t < 8; ++t)
      crow[t * 16 + l15] = f2bf(o[t][r] * inv);
  }
}

// ---------------------------------------------------------------------------
extern "C" void kernel_launch(void* const* d_in, const int* in_sizes, int n_in,
                              void* d_out, int out_size, void* d_ws, size_t ws_size,
                              hipStream_t stream) {
  const float* x      = (const float*)d_in[0];
  const float* past_K = (const float*)d_in[1];
  const float* past_V = (const float*)d_in[2];
  const float* Wq = (const float*)d_in[3];
  const float* bq = (const float*)d_in[4];
  const float* Wk = (const float*)d_in[5];
  const float* bk = (const float*)d_in[6];
  const float* Wv = (const float*)d_in[7];
  const float* bv = (const float*)d_in[8];
  const float* Wo = (const float*)d_in[9];
  const float* bo = (const float*)d_in[10];

  float* out  = (float*)d_out;                       // [2,2048,2048]
  float* Kout = out + (size_t)SB * SS * D_MODEL;     // fp32 cache [2,16,4096,128]
  float* Vout = Kout + (size_t)SB * NH * TT * DH;

  __bf16* Qbf   = (__bf16*)d_ws;                            // [4096,2048]
  __bf16* CtxBf = Qbf + (size_t)SB * SS * D_MODEL;          // [4096,2048]
  __bf16* Kbf   = CtxBf + (size_t)SB * SS * D_MODEL;        // [32,4096,128]
  __bf16* VbfT  = Kbf + (size_t)SB * NH * TT * DH;          // [32,128,4096]

  // 1) past K/V -> fp32 cache front + bf16 staging
  copy_past<<<8192, 256, 0, stream>>>((const v4f*)past_K, (const v4f*)past_V,
                                      (v4f*)Kout, (v4f*)Vout, Kbf, VbfT);

  // 2) Q/K/V projections
  dim3 gg(D_MODEL / 128, (SB * SS) / 128);
  gemm_wmma<0><<<gg, 256, 0, stream>>>(x, Wq, bq, nullptr, Qbf, 1);
  gemm_wmma<0><<<gg, 256, 0, stream>>>(x, Wk, bk, Kout, Kbf, 2);
  gemm_wmma<0><<<gg, 256, 0, stream>>>(x, Wv, bv, Vout, VbfT, 3);

  // 3) attention over full cache (bf16 staging buffers)
  attn_wmma<<<dim3(SS / 128, SB * NH), 256, 0, stream>>>(Qbf, Kbf, VbfT, CtxBf);

  // 4) output projection from bf16 ctx
  gemm_wmma<1><<<gg, 256, 0, stream>>>(CtxBf, Wo, bo, out, nullptr, 0);
}
